// CoarseMatching_22222160789775
// MI455X (gfx1250) — matrix operand order repaired and checked by reading, not compiled, not run on hardware
//
#include <hip/hip_runtime.h>
#include <hip/hip_bf16.h>
#include <math.h>

typedef __attribute__((ext_vector_type(16))) _Float16 v16h;
typedef __attribute__((ext_vector_type(8)))  float    v8f;

#define NB    2
#define LROWS 4800   // 60*80
#define SCOLS 4800   // 60*80
#define CDIM  256
#define H0C   60
#define W0C   80
#define NEGINF (-1000000000.0f)
// (1/sqrt(C))*(1/sqrt(C))/TEMP = (1/256)/0.1
#define SIM_SCALE 0.0390625f

// ---------------------------------------------------------------------------
// K0: effective heights/widths from the padded masks (tiny, one block)
// dims[0..1]=hs0, dims[2..3]=ws0, dims[4..5]=hs1, dims[6..7]=ws1
// ---------------------------------------------------------------------------
__global__ void k0_dims(const int* __restrict__ mask0,
                        const int* __restrict__ mask1,
                        int* __restrict__ dims) {
  __shared__ int buf[128];
  const int t = threadIdx.x;
  for (int which = 0; which < 2; ++which) {
    const int* mk = which == 0 ? mask0 : mask1;
    for (int n = 0; n < NB; ++n) {
      const int* m = mk + n * (H0C * W0C);
      // hs = max over w of (sum over h)
      int colsum = 0;
      if (t < W0C) {
        for (int h = 0; h < H0C; ++h) colsum += (m[h * W0C + t] != 0);
        buf[t] = colsum;
      }
      __syncthreads();
      if (t == 0) {
        int mx = 0;
        for (int i = 0; i < W0C; ++i) mx = max(mx, buf[i]);
        dims[which * 4 + n] = mx;
      }
      __syncthreads();
      // ws = max over h of (sum over w)
      int rowsum = 0;
      if (t < H0C) {
        for (int w = 0; w < W0C; ++w) rowsum += (m[t * W0C + w] != 0);
        buf[t] = rowsum;
      }
      __syncthreads();
      if (t == 0) {
        int mx = 0;
        for (int i = 0; i < H0C; ++i) mx = max(mx, buf[i]);
        dims[which * 4 + 2 + n] = mx;
      }
      __syncthreads();
    }
  }
}

// ---------------------------------------------------------------------------
// K1: WMMA GEMM  sim[n,l,s] = (f0[n,l,:] . f1[n,s,:]) * SIM_SCALE, masked.
// One wave computes a 32x64 strip: 2 A-fragments x 4 B-fragments = 8
// accumulators. Each B fragment feeds two back-to-back WMMAs (no hazard
// NOPs between them, since D does not overlap the next A/B), halving the
// load+convert work per WMMA versus a 16x64 tile.
// ---------------------------------------------------------------------------
__global__ void __launch_bounds__(256)
k1_gemm_sim(const float* __restrict__ f0,
            const float* __restrict__ f1,
            const int* __restrict__ mask0,
            const int* __restrict__ mask1,
            float* __restrict__ sim) {
  const int lane    = threadIdx.x & 31;
  const int wave    = threadIdx.x >> 5;
  const int mtilesN = LROWS / 32;            // 150
  const int ntilesN = SCOLS / 64;            // 75
  const int tilesPerBatch = mtilesN * ntilesN;   // 11250
  long wid = (long)blockIdx.x * 8 + wave;
  if (wid >= (long)NB * tilesPerBatch) return;
  const int n     = (int)(wid / tilesPerBatch);
  const int rem   = (int)(wid % tilesPerBatch);
  const int mtile = rem / ntilesN;
  const int ntile = rem % ntilesN;

  const float* A = f0 + (size_t)n * LROWS * CDIM;
  const float* B = f1 + (size_t)n * SCOLS * CDIM;

  const int lidx    = lane & 15;
  const int halfSel = lane >> 4;

  v8f acc[2][4];
  v8f zero = {};
#pragma unroll
  for (int mi = 0; mi < 2; ++mi)
#pragma unroll
    for (int j = 0; j < 4; ++j) acc[mi][j] = zero;

  // per-lane base pointers (row-major contiguous runs -> global_load_b128)
  const float* aRow0 = A + (size_t)(mtile * 32 + lidx) * CDIM;        // M 0..15
  const float* aRow1 = A + (size_t)(mtile * 32 + 16 + lidx) * CDIM;   // M 16..31
  const float* bCol[4];
#pragma unroll
  for (int j = 0; j < 4; ++j)
    bCol[j] = B + (size_t)(ntile * 64 + j * 16 + lidx) * CDIM;

  for (int k0 = 0; k0 < CDIM; k0 += 32) {
    // prefetch the next K-step's lines (global_prefetch_b8); cheap, helps the
    // cold first pass before features settle into L2.
    if (k0 + 32 < CDIM) {
      __builtin_prefetch(aRow0 + k0 + 32, 0, 3);
      __builtin_prefetch(aRow1 + k0 + 32, 0, 3);
#pragma unroll
      for (int j = 0; j < 4; ++j) __builtin_prefetch(bCol[j] + k0 + 32, 0, 3);
    }

    // ----- A fragments (16x32 f16 layout): lanes 0-15 hold K 0..7 & 16..23,
    //       lanes 16-31 hold K 8..15 & 24..31, for row M = lane&15
    v16h a0, a1;
    {
      const float* ap0 = aRow0 + k0 + halfSel * 8;
      const float* ap1 = aRow1 + k0 + halfSel * 8;
#pragma unroll
      for (int i = 0; i < 8; ++i) a0[i] = (_Float16)ap0[i];
#pragma unroll
      for (int i = 0; i < 8; ++i) a0[8 + i] = (_Float16)ap0[16 + i];
#pragma unroll
      for (int i = 0; i < 8; ++i) a1[i] = (_Float16)ap1[i];
#pragma unroll
      for (int i = 0; i < 8; ++i) a1[8 + i] = (_Float16)ap1[16 + i];
    }
    const int kb2 = k0 + halfSel * 8;    // B: lanes 0-15 hold K 0..7 & 16..23, lanes 16-31 K 8..15 & 24..31
#pragma unroll
    for (int j = 0; j < 4; ++j) {
      const float* bp = bCol[j] + kb2;
      v16h b;
#pragma unroll
      for (int i = 0; i < 16; ++i) b[i] = (_Float16)bp[(i < 8) ? i : (i + 8)];
      acc[0][j] = __builtin_amdgcn_wmma_f32_16x16x32_f16(
          false, a0, false, b, (short)0, acc[0][j], false, false);
      acc[1][j] = __builtin_amdgcn_wmma_f32_16x16x32_f16(
          false, a1, false, b, (short)0, acc[1][j], false, false);
    }
  }

  // ----- epilogue: scale + mask.  D layout: VGPR r -> row r (lanes 0-15)
  //       or row r+8 (lanes 16-31), col = lane&15.
  int rmask[2][8];
#pragma unroll
  for (int mi = 0; mi < 2; ++mi)
#pragma unroll
    for (int r = 0; r < 8; ++r)
      rmask[mi][r] = mask0[n * LROWS + mtile * 32 + mi * 16 + r + halfSel * 8];

  float* out = sim + (size_t)n * LROWS * SCOLS;
#pragma unroll
  for (int j = 0; j < 4; ++j) {
    const int col = ntile * 64 + j * 16 + lidx;
    const int m1  = mask1[n * SCOLS + col];
#pragma unroll
    for (int mi = 0; mi < 2; ++mi) {
#pragma unroll
      for (int r = 0; r < 8; ++r) {
        const int m = mtile * 32 + mi * 16 + r + halfSel * 8;
        const float v = acc[mi][j][r] * SIM_SCALE;
        out[(size_t)m * SCOLS + col] = (m1 && rmask[mi][r]) ? v : NEGINF;
      }
    }
  }
}

// ---------------------------------------------------------------------------
// helpers: online-softmax update
// ---------------------------------------------------------------------------
__device__ inline void sm_update(float& m, float& d, float x) {
  float nm = fmaxf(m, x);
  d = d * __expf(m - nm) + __expf(x - nm);
  m = nm;
}

// K2: per-row (axis=2) softmax stats: rmax[N*L], rsum[N*L]
__global__ void k2_row_stats(const float* __restrict__ sim,
                             float* __restrict__ rmax,
                             float* __restrict__ rsum) {
  const int row = blockIdx.x;
  const float* p = sim + (size_t)row * SCOLS;
  float m = -3.4e38f, d = 0.0f;
  for (int s = threadIdx.x; s < SCOLS; s += 256) sm_update(m, d, p[s]);
  __shared__ float smx[256], sdd[256];
  smx[threadIdx.x] = m; sdd[threadIdx.x] = d;
  __syncthreads();
  for (int off = 128; off > 0; off >>= 1) {
    if (threadIdx.x < off) {
      float m2 = smx[threadIdx.x + off], d2 = sdd[threadIdx.x + off];
      float M = fmaxf(smx[threadIdx.x], m2);
      sdd[threadIdx.x] = sdd[threadIdx.x] * __expf(smx[threadIdx.x] - M) + d2 * __expf(m2 - M);
      smx[threadIdx.x] = M;
    }
    __syncthreads();
  }
  if (threadIdx.x == 0) { rmax[row] = smx[0]; rsum[row] = sdd[0]; }
}

// K3: per-column (axis=1) softmax stats: cmax[N*S], csum[N*S]
__global__ void k3_col_stats(const float* __restrict__ sim,
                             float* __restrict__ cmax,
                             float* __restrict__ csum) {
  const int n = blockIdx.y;
  const int s = blockIdx.x * 256 + threadIdx.x;
  if (s >= SCOLS) return;
  const float* p = sim + (size_t)n * LROWS * SCOLS + s;
  float m = -3.4e38f, d = 0.0f;
  for (int l = 0; l < LROWS; ++l) sm_update(m, d, p[(size_t)l * SCOLS]);
  cmax[n * SCOLS + s] = m;
  csum[n * SCOLS + s] = d;
}

// K4: conf = softmax_row * softmax_col, in place; also per-row conf max
__global__ void k4_dual_softmax(float* __restrict__ conf,
                                const float* __restrict__ rmax,
                                const float* __restrict__ rsum,
                                const float* __restrict__ cmax,
                                const float* __restrict__ csum,
                                float* __restrict__ rowmaxc) {
  const int row = blockIdx.x;
  const int n = row / LROWS;
  const float rm = rmax[row];
  const float rinv = 1.0f / rsum[row];
  float* p = conf + (size_t)row * SCOLS;
  const float* cm = cmax + n * SCOLS;
  const float* cd = csum + n * SCOLS;
  float best = 0.0f;
  for (int s = threadIdx.x; s < SCOLS; s += 256) {
    float x = p[s];
    float v = (__expf(x - rm) * rinv) * (__expf(x - cm[s]) / cd[s]);
    p[s] = v;
    best = fmaxf(best, v);
  }
  __shared__ float sb[256];
  sb[threadIdx.x] = best;
  __syncthreads();
  for (int off = 128; off > 0; off >>= 1) {
    if (threadIdx.x < off) sb[threadIdx.x] = fmaxf(sb[threadIdx.x], sb[threadIdx.x + off]);
    __syncthreads();
  }
  if (threadIdx.x == 0) rowmaxc[row] = sb[0];
}

// K5: per-column conf max
__global__ void k5_col_max(const float* __restrict__ conf,
                           float* __restrict__ colmaxc) {
  const int n = blockIdx.y;
  const int s = blockIdx.x * 256 + threadIdx.x;
  if (s >= SCOLS) return;
  const float* p = conf + (size_t)n * LROWS * SCOLS + s;
  float m = 0.0f;
  for (int l = 0; l < LROWS; ++l) m = fmaxf(m, p[(size_t)l * SCOLS]);
  colmaxc[n * SCOLS + s] = m;
}

// K6: threshold + border + mutual-NN + dense match extraction
__global__ void k6_extract(const float* __restrict__ conf,
                           const float* __restrict__ rowmaxc,
                           const float* __restrict__ colmaxc,
                           const int* __restrict__ dims,
                           float* __restrict__ maskv,
                           float* __restrict__ jids,
                           float* __restrict__ mconf,
                           float* __restrict__ mk0,
                           float* __restrict__ mk1) {
  const int row = blockIdx.x;
  const int n = row / LROWS;
  const int l = row % LROWS;
  const int h0 = l / W0C, w0 = l % W0C;
  const int hs0 = dims[0 + n], ws0 = dims[2 + n];
  const int hs1 = dims[4 + n], ws1 = dims[6 + n];
  const bool v0 = (h0 >= 2) && (h0 < hs0 - 2) && (w0 >= 2) && (w0 < ws0 - 2);
  const float rmc = rowmaxc[row];
  const float* p = conf + (size_t)row * SCOLS;
  const float* cmc = colmaxc + n * SCOLS;
  int best = SCOLS;
  if (v0) {
    for (int s = threadIdx.x; s < SCOLS; s += 256) {
      float c = p[s];
      if (c > 0.2f && c == rmc && c == cmc[s]) {
        int h1 = s / W0C, w1 = s % W0C;
        if (h1 >= 2 && h1 < hs1 - 2 && w1 >= 2 && w1 < ws1 - 2)
          best = (s < best) ? s : best;
      }
    }
  }
  __shared__ int sb[256];
  sb[threadIdx.x] = best;
  __syncthreads();
  for (int off = 128; off > 0; off >>= 1) {
    if (threadIdx.x < off) {
      int o = sb[threadIdx.x + off];
      if (o < sb[threadIdx.x]) sb[threadIdx.x] = o;
    }
    __syncthreads();
  }
  if (threadIdx.x == 0) {
    int j = sb[0];
    const bool found = j < SCOLS;
    const int jj = found ? j : 0;           // argmax of all-false == 0
    maskv[row] = found ? 1.0f : 0.0f;
    jids[row]  = (float)jj;
    mconf[row] = found ? p[jj] : 0.0f;
    mk0[(size_t)row * 2 + 0] = (float)(w0 * 8);
    mk0[(size_t)row * 2 + 1] = (float)(h0 * 8);
    mk1[(size_t)row * 2 + 0] = (float)((jj % W0C) * 8);
    mk1[(size_t)row * 2 + 1] = (float)((jj / W0C) * 8);
  }
}

// ---------------------------------------------------------------------------
extern "C" void kernel_launch(void* const* d_in, const int* in_sizes, int n_in,
                              void* d_out, int out_size, void* d_ws, size_t ws_size,
                              hipStream_t stream) {
  const float* f0   = (const float*)d_in[0];   // [N, L, C]
  const float* f1   = (const float*)d_in[1];   // [N, S, C]
  const int*  mask0 = (const int*)d_in[2];     // [N, 60, 80]
  const int*  mask1 = (const int*)d_in[3];     // [N, 60, 80]

  // d_out layout (floats, return order): conf, mask_v, j_ids, mconf, mkpts0, mkpts1
  float* out   = (float*)d_out;
  float* conf  = out;
  float* maskv = conf  + (size_t)NB * LROWS * SCOLS;
  float* jids  = maskv + (size_t)NB * LROWS;
  float* mconf = jids  + (size_t)NB * LROWS;
  float* mk0   = mconf + (size_t)NB * LROWS;
  float* mk1   = mk0   + (size_t)NB * LROWS * 2;

  // workspace
  int*   dims    = (int*)d_ws;                 // 8 ints
  float* wsf     = (float*)d_ws + 16;
  float* rmax    = wsf;                        // N*L
  float* rsum    = rmax    + NB * LROWS;
  float* cmax    = rsum    + NB * LROWS;       // N*S
  float* csum    = cmax    + NB * SCOLS;
  float* rowmaxc = csum    + NB * SCOLS;       // N*L
  float* colmaxc = rowmaxc + NB * LROWS;       // N*S

  k0_dims<<<1, 256, 0, stream>>>(mask0, mask1, dims);

  const int wavesTotal = NB * (LROWS / 32) * (SCOLS / 64);   // 22500
  const int gemmBlocks = (wavesTotal + 7) / 8;               // 2813
  k1_gemm_sim<<<gemmBlocks, 256, 0, stream>>>(f0, f1, mask0, mask1, conf);

  k2_row_stats<<<NB * LROWS, 256, 0, stream>>>(conf, rmax, rsum);

  dim3 colGrid((SCOLS + 255) / 256, NB);
  k3_col_stats<<<colGrid, 256, 0, stream>>>(conf, cmax, csum);

  k4_dual_softmax<<<NB * LROWS, 256, 0, stream>>>(conf, rmax, rsum, cmax, csum, rowmaxc);

  k5_col_max<<<colGrid, 256, 0, stream>>>(conf, colmaxc);

  k6_extract<<<NB * LROWS, 256, 0, stream>>>(conf, rowmaxc, colmaxc, dims,
                                             maskv, jids, mconf, mk0, mk1);
}
